// FedformerFourierBase_74912819577316
// MI455X (gfx1250) — matrix operand order repaired and checked by reading, not compiled, not run on hardware
//
#include <hip/hip_runtime.h>
#include <hip/hip_bf16.h>
#include <math.h>

// ---------------------------------------------------------------------------
// Fedformer Fourier block: rfft(64 modes) -> complex mode mixing -> irfft.
// B=32, L=4096, H=8, E=64, MODES=64.  Only q / w_re / w_im are used.
//
// Three dense matmul stages, all on v_wmma_f32_16x16x32_f16 (f16 in, f32 acc):
//   1) R(128x64)   = F^T(128x4096) * x^T(4096x64)      per (b,h)   ~17.2 GF
//   2) Y           = complex mix, four 32x64x64 GEMMs   per (h,m)  ~0.54 GF
//   3) out(64x4096)= Y(64x128) * G^T                    per (b,h)  ~17.2 GF
//
// Kernel 1 uses a two-deep software pipeline: global loads for chunk k+1 are
// issued at the END of iteration k-1 and only consumed (convert + LDS
// transpose stores) at the END of iteration k, one __syncthreads() later.
// The barrier is a scheduling fence, so the loads get a full iteration of
// WMMA work to cover their latency, and s_wait_dscnt before the WMMAs only
// covers the fragment ds_loads.
//
// Workspace layout (14 MB total):
//   [0,1MB)   Fh  : f16 forward basis, F[t][n], t<4096, n<128
//   [1,2MB)   Gth : f16 inverse basis, G[t][n] = a(n%64)*F[t][n]
//   [2,10MB)  xm  : f32 truncated spectrum, xm[b*H+h][n][i]  (256*128*64)
//   [10,14MB) ym  : f16 mixed spectrum,     ym[b*H+h][o][n]  (256*64*128)
// ---------------------------------------------------------------------------

typedef __attribute__((ext_vector_type(16))) _Float16 v16h;
typedef __attribute__((ext_vector_type(8)))  _Float16 v8h;
typedef __attribute__((ext_vector_type(8)))  float    v8f;

#define B_  32
#define L_  4096
#define H_  8
#define E_  64
#define MODES_ 64
#define NF  128   // 2*MODES (re | im)
#define NCH (L_ / 32)

union Frag { v16h v; v8h h[2]; };

// ---------------------------------------------------------------------------
// Kernel 0: build f16 DFT bases.
//   F[t][n<64]  =  cos(2*pi*n*t/L)         (pairs with Re X)
//   F[t][64+m]  = -sin(2*pi*m*t/L)         (pairs with Im X)
//   G[t][n]     = a * F[t][n],  a = (mode==0 ? 1 : 2)/L   (irfft weights)
// ---------------------------------------------------------------------------
__global__ __launch_bounds__(256)
void fed_basis_kernel(_Float16* __restrict__ Fh, _Float16* __restrict__ Gth) {
    int idx  = blockIdx.x * 256 + threadIdx.x;     // t*128 + n
    int t    = idx >> 7;
    int n    = idx & 127;
    int mode = n & 63;
    float th = 6.2831853071795864769f * (float)mode * (float)t / (float)L_;
    float s, c;
    __sincosf(th, &s, &c);
    float f = (n < 64) ? c : -s;
    float a = (mode == 0 ? 1.0f : 2.0f) / (float)L_;
    Fh[idx]  = (_Float16)f;
    Gth[idx] = (_Float16)(a * f);
}

// ---------------------------------------------------------------------------
// Kernel 1: truncated forward DFT, one workgroup per (b,h).
// K marched in chunks of 32 with double-buffered LDS; tiles stored transposed
// ([n][t], [i][t]) so every WMMA fragment is two contiguous ds_load_b128.
// ---------------------------------------------------------------------------
struct ChunkRegs { v8h f0, f1; float4 x0, x1; };

__global__ __launch_bounds__(256)
void fed_dft_kernel(const float* __restrict__ q,
                    const _Float16* __restrict__ Fh,
                    float* __restrict__ xm) {
    __shared__ _Float16 ldsF[2][NF][32];  // [buf][n][t]  16 KB
    __shared__ _Float16 ldsX[2][E_][32];  // [buf][i][t]   8 KB

    const int bh   = blockIdx.x;          // b*H + h
    const int b    = bh >> 3;
    const int h    = bh & 7;
    const int tid  = threadIdx.x;
    const int lane = tid & 31;
    const int w    = tid >> 5;            // wave 0..7 -> n-tile
    const int hi   = lane >> 4;           // half-wave select
    const int ln   = lane & 15;

    const float* qbase = q + ((size_t)b * L_ * H_ + h) * E_;  // + t*512 + i

    // cooperative loader mapping
    const int lt   = tid >> 3;            // t within chunk (0..31)
    const int li8  = (tid & 7) * 8;       // i0 for x (8 floats)
    const int ln16 = (tid & 7) * 16;      // n0 for F (16 halves)

    // issue global loads for one K-chunk (registers only; LOADcnt)
    auto issue_loads = [&](int kc) {
        ChunkRegs r;
        const v8h* fsrc = (const v8h*)(Fh + (size_t)(kc + lt) * NF + ln16);
        r.f0 = fsrc[0];
        r.f1 = fsrc[1];
        const float4* xs = (const float4*)(qbase + (size_t)(kc + lt) * (H_ * E_) + li8);
        r.x0 = xs[0];
        r.x1 = xs[1];
        return r;
    };
    // convert + transpose-store into LDS (DScnt)
    auto commit_chunk = [&](const ChunkRegs& r, int buf) {
#pragma unroll
        for (int j = 0; j < 8; ++j) ldsF[buf][ln16 + j][lt]     = r.f0[j];
#pragma unroll
        for (int j = 0; j < 8; ++j) ldsF[buf][ln16 + 8 + j][lt] = r.f1[j];
        _Float16 xh[8] = {(_Float16)r.x0.x, (_Float16)r.x0.y, (_Float16)r.x0.z, (_Float16)r.x0.w,
                          (_Float16)r.x1.x, (_Float16)r.x1.y, (_Float16)r.x1.z, (_Float16)r.x1.w};
#pragma unroll
        for (int j = 0; j < 8; ++j) ldsX[buf][li8 + j][lt] = xh[j];
    };

    v8f acc[4] = {};

    // ---- pipeline prologue: chunk 0 committed, chunk 1 in flight ----
    {
        ChunkRegs c0 = issue_loads(0);
        commit_chunk(c0, 0);
    }
    ChunkRegs pend = issue_loads(32);

    const int nrow = 16 * w + ln;         // A-matrix row (n) for this lane

    for (int kci = 0; kci < NCH; ++kci) {
        __syncthreads();
        const int cur = kci & 1;

        // fragment DS loads, then WMMAs (DS waits cover only these loads)
        Frag aF;
        aF.h[0] = *(const v8h*)&ldsF[cur][nrow][hi * 16];
        aF.h[1] = *(const v8h*)&ldsF[cur][nrow][hi * 16 + 8];
        Frag bX[4];
#pragma unroll
        for (int it = 0; it < 4; ++it) {
            const int irow = 16 * it + ln;
            bX[it].h[0] = *(const v8h*)&ldsX[cur][irow][hi * 16];
            bX[it].h[1] = *(const v8h*)&ldsX[cur][irow][hi * 16 + 8];
        }
#pragma unroll
        for (int it = 0; it < 4; ++it)
            acc[it] = __builtin_amdgcn_wmma_f32_16x16x32_f16(
                false, aF.v, false, bX[it].v, (short)0, acc[it], false, false);

        // commit the chunk whose loads were issued one iteration ago, then
        // put the next chunk's loads in flight across the coming barrier
        if (kci + 1 < NCH) {
            commit_chunk(pend, cur ^ 1);
            if (kci + 2 < NCH) {
                __builtin_prefetch(qbase + (size_t)(32 * (kci + 3) + lt) * (H_ * E_) + li8, 0, 0);
                pend = issue_loads(32 * (kci + 2));
            }
        }
    }

    // D layout: row n = 16w + v + 8*hi, col i = 16*it + ln
    float* xout = xm + (size_t)bh * NF * E_;
#pragma unroll
    for (int it = 0; it < 4; ++it) {
#pragma unroll
        for (int v = 0; v < 8; ++v) {
            int n = 16 * w + v + hi * 8;
            int i = 16 * it + ln;
            xout[n * E_ + i] = acc[it][v];
        }
    }
}

// ---------------------------------------------------------------------------
// Kernel 2: complex mode mixing.  One workgroup per (h,m):
//   Yre = Xr*Wr - Xi*Wi      Yim = Xr*Wi + Xi*Wr
// (f16 WMMA has no A/B negation on CDNA5, so -Xi is formed in VALU.)
// ---------------------------------------------------------------------------
__global__ __launch_bounds__(256)
void fed_mix_kernel(const float* __restrict__ xm,
                    const float* __restrict__ w_re,
                    const float* __restrict__ w_im,
                    _Float16* __restrict__ ym) {
    __shared__ _Float16 ldsXr[B_][E_];   // [b][i]
    __shared__ _Float16 ldsXi[B_][E_];
    __shared__ _Float16 ldsWr[E_][E_];   // [o][i] (transposed for B frags)
    __shared__ _Float16 ldsWi[E_][E_];

    const int hm   = blockIdx.x;         // h*64 + m
    const int h    = hm >> 6;
    const int m    = hm & 63;
    const int tid  = threadIdx.x;
    const int lane = tid & 31;
    const int w    = tid >> 5;
    const int hi   = lane >> 4;
    const int ln   = lane & 15;

    {   // load X (re, im): rows contiguous in i
        int bb = tid >> 3;
        int i0 = (tid & 7) * 8;
        const float4* sr = (const float4*)(xm + (((size_t)(bb * H_ + h) * NF) + m) * E_ + i0);
        const float4* si = (const float4*)(xm + (((size_t)(bb * H_ + h) * NF) + 64 + m) * E_ + i0);
        float4 r0 = sr[0], r1 = sr[1];
        float4 i0v = si[0], i1v = si[1];
        _Float16 rh[8] = {(_Float16)r0.x, (_Float16)r0.y, (_Float16)r0.z, (_Float16)r0.w,
                          (_Float16)r1.x, (_Float16)r1.y, (_Float16)r1.z, (_Float16)r1.w};
        _Float16 ih[8] = {(_Float16)i0v.x, (_Float16)i0v.y, (_Float16)i0v.z, (_Float16)i0v.w,
                          (_Float16)i1v.x, (_Float16)i1v.y, (_Float16)i1v.z, (_Float16)i1v.w};
#pragma unroll
        for (int j = 0; j < 8; ++j) { ldsXr[bb][i0 + j] = rh[j]; ldsXi[bb][i0 + j] = ih[j]; }
    }
    {   // load W[h,i,o,m] (stride 64 over o) -> transposed LDS [o][i]
        int i  = tid >> 2;
        int o0 = (tid & 3) * 16;
        const float* wr = w_re + (((size_t)(h * E_ + i) * E_) + o0) * MODES_ + m;
        const float* wi = w_im + (((size_t)(h * E_ + i) * E_) + o0) * MODES_ + m;
        float vr[16], vi[16];
#pragma unroll
        for (int j = 0; j < 16; ++j) { vr[j] = wr[(size_t)j * MODES_]; vi[j] = wi[(size_t)j * MODES_]; }
#pragma unroll
        for (int j = 0; j < 16; ++j) {
            ldsWr[o0 + j][i] = (_Float16)vr[j];
            ldsWi[o0 + j][i] = (_Float16)vi[j];
        }
    }
    __syncthreads();

    const int comp = w & 1;              // 0 = Yre, 1 = Yim
    const int mt   = (w >> 1) & 1;       // b tile
    const int nt0  = (w >> 2) * 2;       // first o tile of two

    v8f acc[2] = {};
    const int brow = 16 * mt + ln;
#pragma unroll
    for (int kk = 0; kk < 2; ++kk) {
        Frag aR, aI;
        aR.h[0] = *(const v8h*)&ldsXr[brow][kk * 32 + hi * 16];
        aR.h[1] = *(const v8h*)&ldsXr[brow][kk * 32 + hi * 16 + 8];
        aI.h[0] = *(const v8h*)&ldsXi[brow][kk * 32 + hi * 16];
        aI.h[1] = *(const v8h*)&ldsXi[brow][kk * 32 + hi * 16 + 8];
        Frag bR[2], bI[2];
#pragma unroll
        for (int j = 0; j < 2; ++j) {
            const int orow = 16 * (nt0 + j) + ln;
            bR[j].h[0] = *(const v8h*)&ldsWr[orow][kk * 32 + hi * 16];
            bR[j].h[1] = *(const v8h*)&ldsWr[orow][kk * 32 + hi * 16 + 8];
            bI[j].h[0] = *(const v8h*)&ldsWi[orow][kk * 32 + hi * 16];
            bI[j].h[1] = *(const v8h*)&ldsWi[orow][kk * 32 + hi * 16 + 8];
        }
        v16h aI2 = comp ? aI.v : -aI.v;  // Yre needs -Xi*Wi
#pragma unroll
        for (int j = 0; j < 2; ++j) {
            v16h b1 = comp ? bI[j].v : bR[j].v;
            v16h b2 = comp ? bR[j].v : bI[j].v;
            acc[j] = __builtin_amdgcn_wmma_f32_16x16x32_f16(
                false, aR.v, false, b1, (short)0, acc[j], false, false);
            acc[j] = __builtin_amdgcn_wmma_f32_16x16x32_f16(
                false, aI2, false, b2, (short)0, acc[j], false, false);
        }
    }
    // ym[bh][o][comp*64 + m]
#pragma unroll
    for (int j = 0; j < 2; ++j) {
#pragma unroll
        for (int v = 0; v < 8; ++v) {
            int bb = 16 * mt + v + hi * 8;
            int o  = 16 * (nt0 + j) + ln;
            ym[((size_t)(bb * H_ + h) * E_ + o) * NF + comp * 64 + m] = (_Float16)acc[j][v];
        }
    }
}

// ---------------------------------------------------------------------------
// Kernel 3: inverse DFT.  Grid = (b,h) x 16 column segments; each workgroup
// computes a 64x256 output tile: out(64o x 256t) = Y(64x128) * G^T.
// Y staged once in LDS; all 8 B fragments of a K-step are loaded from the
// L2-resident basis before the 8 WMMAs, so L2 latency is paid once per step
// (4096 workgroups give ample cross-WG latency hiding).
// ---------------------------------------------------------------------------
__global__ __launch_bounds__(256)
void fed_idft_kernel(const _Float16* __restrict__ ymh,
                     const _Float16* __restrict__ Gth,
                     float* __restrict__ out) {
    __shared__ _Float16 ldsY[E_][NF];    // 16 KB

    const int bh   = blockIdx.x >> 4;
    const int tseg = blockIdx.x & 15;
    const int tid  = threadIdx.x;
    const int lane = tid & 31;
    const int w    = tid >> 5;
    const int hi   = lane >> 4;
    const int ln   = lane & 15;

    {   // linear copy of Y (8192 halves)
        const v8h* src = (const v8h*)(ymh + (size_t)bh * E_ * NF) + tid * 4;
        v8h* dst = (v8h*)&ldsY[0][0] + tid * 4;
#pragma unroll
        for (int j = 0; j < 4; ++j) dst[j] = src[j];
    }
    __syncthreads();

    const int mt    = w & 3;             // o tile
    const int th    = w >> 2;            // column half
    const int tbase = tseg * 256 + th * 128;

    v8f acc[8] = {};
    const int orow = 16 * mt + ln;
#pragma unroll
    for (int kk = 0; kk < 4; ++kk) {
        // issue all loads of this K-step first, then the 8 WMMAs
        Frag bG[8];
#pragma unroll
        for (int j = 0; j < 8; ++j) {
            const int t = tbase + 16 * j + ln;
            const v8h* g = (const v8h*)(Gth + (size_t)t * NF + kk * 32 + hi * 16);
            bG[j].h[0] = g[0];
            bG[j].h[1] = g[1];
        }
        Frag aY;
        aY.h[0] = *(const v8h*)&ldsY[orow][kk * 32 + hi * 16];
        aY.h[1] = *(const v8h*)&ldsY[orow][kk * 32 + hi * 16 + 8];
#pragma unroll
        for (int j = 0; j < 8; ++j)
            acc[j] = __builtin_amdgcn_wmma_f32_16x16x32_f16(
                false, aY.v, false, bG[j].v, (short)0, acc[j], false, false);
    }
    float* obase = out + (size_t)bh * E_ * L_;
#pragma unroll
    for (int j = 0; j < 8; ++j) {
        const int t = tbase + 16 * j + ln;
#pragma unroll
        for (int v = 0; v < 8; ++v) {
            int o = 16 * mt + v + hi * 8;
            obase[(size_t)o * L_ + t] = acc[j][v];
        }
    }
}

// ---------------------------------------------------------------------------
extern "C" void kernel_launch(void* const* d_in, const int* in_sizes, int n_in,
                              void* d_out, int out_size, void* d_ws, size_t ws_size,
                              hipStream_t stream) {
    const float* q    = (const float*)d_in[0];
    // d_in[1] = k, d_in[2] = v : unused by the reference computation
    const float* w_re = (const float*)d_in[3];
    const float* w_im = (const float*)d_in[4];
    float* out = (float*)d_out;

    char* ws = (char*)d_ws;                       // requires >= 14 MB
    _Float16* Fh  = (_Float16*)(ws);              // 1 MB
    _Float16* Gth = (_Float16*)(ws + (1u << 20)); // 1 MB
    float*    xm  = (float*)   (ws + (2u << 20)); // 8 MB
    _Float16* ym  = (_Float16*)(ws + (10u << 20));// 4 MB

    fed_basis_kernel<<<(L_ * NF) / 256, 256, 0, stream>>>(Fh, Gth);
    fed_dft_kernel  <<<B_ * H_,          256, 0, stream>>>(q, Fh, xm);
    fed_mix_kernel  <<<H_ * MODES_,      256, 0, stream>>>(xm, w_re, w_im, ym);
    fed_idft_kernel <<<B_ * H_ * 16,     256, 0, stream>>>(ym, Gth, out);
}
